// SharedExpertMoE_52888227283708
// MI455X (gfx1250) — matrix operand order repaired and checked by reading, not compiled, not run on hardware
//
#include <hip/hip_runtime.h>
#include <hip/hip_bf16.h>
#include <math.h>

// ---------------- problem constants ----------------
#define DM   1024          // d_model
#define DF   2048          // d_ffn
#define NE   8             // experts
#define TT   4096          // tokens (2*2048)
#define NSLOT (TT*2)       // top-2 slots

#define KSTEP  32          // K macro-step staged in LDS
#define ROWPAD 40          // LDS row stride in elements (32 + 8 pad -> conflict-free b128)

typedef __bf16 bf16_t;
typedef __attribute__((ext_vector_type(8)))  bf16_t v8bf;
typedef __attribute__((ext_vector_type(16))) bf16_t v16bf;
typedef __attribute__((ext_vector_type(8)))  float  v8f;

// ---------------- CDNA5 primitives ----------------
__device__ __forceinline__ v8f wmma_bf16(v16bf a, v16bf b, v8f c) {
  return __builtin_amdgcn_wmma_f32_16x16x32_bf16(false, a, false, b, (short)0, c, false, false);
}

// LDS offset of a generic pointer to __shared__: flat LDS aperture is
// {SHARED_BASE[63:32], lds_offset[31:0]} (ISA 10.2), so truncation suffices.
__device__ __forceinline__ unsigned lds_off(const void* p) {
  return (unsigned)(unsigned long long)p;
}

// async global->LDS copy, 16B per lane, tracked by ASYNCcnt
__device__ __forceinline__ void async_cp16(unsigned ldsoff, const bf16_t* src) {
  asm volatile("global_load_async_to_lds_b128 %0, %1, off"
               :: "v"(ldsoff), "v"(src) : "memory");
}
#define WAIT_ASYNC(n) asm volatile("s_wait_asynccnt %0" :: "n"(n) : "memory")

__device__ __forceinline__ float silu(float g) {
  return g / (1.0f + __expf(-g));
}

// A fragment from LDS tile (row-major, ROWPAD stride); ISA 16-bit A layout:
// lane<16: K {0..7,16..23}; lane>=16: K {8..15,24..31}
__device__ __forceinline__ v16bf lds_a(const bf16_t* A, int row, int lane) {
  const int half = lane >> 4;
  const bf16_t* p = A + row * ROWPAD + half * 8;
  union { v16bf v; v8bf h[2]; } u;
  u.h[0] = *(const v8bf*)(p);
  u.h[1] = *(const v8bf*)(p + 16);
  return u.v;
}
// B fragment: lane n reads K {0..7,16..23} (lane<16) / {8..15,24..31} (lane>=16) from row n of transposed-weight tile
__device__ __forceinline__ v16bf lds_b(const bf16_t* B, int row, int lane) {
  const int half = lane >> 4;
  const bf16_t* p = B + row * ROWPAD + half * 8;
  union { v16bf v; v8bf h[2]; } u;
  u.h[0] = *(const v8bf*)(p);
  u.h[1] = *(const v8bf*)(p + 16);
  return u.v;
}

// ---------------- elementwise / transpose kernels ----------------
__global__ void moe_cvt_x(const float* __restrict__ x, bf16_t* __restrict__ xb, int n) {
  int i = blockIdx.x * blockDim.x + threadIdx.x;
  if (i < n) xb[i] = (bf16_t)x[i];
}

// src [K,N] f32 (batched along z), dst [N,K] bf16
__global__ __launch_bounds__(256)
void moe_transpose(const float* __restrict__ src, bf16_t* __restrict__ dst, int K, int N) {
  __shared__ float tile[32][33];
  const float* s = src + (size_t)blockIdx.z * K * N;
  bf16_t*      d = dst + (size_t)blockIdx.z * K * N;
  const int kb = blockIdx.x * 32, nb = blockIdx.y * 32;
  const int tx = threadIdx.x & 31, ty = threadIdx.x >> 5;
  for (int r = ty; r < 32; r += 8)
    tile[r][tx] = s[(size_t)(kb + r) * N + nb + tx];
  __syncthreads();
  for (int r = ty; r < 32; r += 8)
    d[(size_t)(nb + r) * K + kb + tx] = (bf16_t)tile[tx][r];
}

// ---------------- router ----------------
__global__ __launch_bounds__(256)
void moe_router(const float* __restrict__ x, const float* __restrict__ rw,
                int* __restrict__ topi, float* __restrict__ topw,
                int* __restrict__ counts, float* __restrict__ psum) {
  const int lane = threadIdx.x & 31;
  const int wave = threadIdx.x >> 5;
  const int t = blockIdx.x * 8 + wave;
  const float* xr = x + (size_t)t * DM;
  float acc[NE];
#pragma unroll
  for (int e = 0; e < NE; ++e) acc[e] = 0.0f;
  for (int d = lane; d < DM; d += 32) {
    const float xv = xr[d];
#pragma unroll
    for (int e = 0; e < NE; ++e) acc[e] += xv * rw[d * NE + e];
  }
#pragma unroll
  for (int e = 0; e < NE; ++e)
    for (int m = 16; m >= 1; m >>= 1) acc[e] += __shfl_xor(acc[e], m, 32);
  if (lane == 0) {
    int i0 = 0;
    for (int e = 1; e < NE; ++e) if (acc[e] > acc[i0]) i0 = e;
    int i1 = (i0 == 0) ? 1 : 0;
    for (int e = 0; e < NE; ++e) if (e != i0 && acc[e] > acc[i1]) i1 = e;
    const float d1 = __expf(acc[i1] - acc[i0]);   // stable 2-way softmax
    const float inv = 1.0f / (1.0f + d1);
    topi[t * 2] = i0; topi[t * 2 + 1] = i1;
    topw[t * 2] = inv; topw[t * 2 + 1] = d1 * inv;
    atomicAdd(&counts[i0], 1);
    atomicAdd(&counts[i1], 1);
    float mx = acc[0];
    for (int e = 1; e < NE; ++e) mx = fmaxf(mx, acc[e]);
    float p[NE], s = 0.0f;
    for (int e = 0; e < NE; ++e) { p[e] = __expf(acc[e] - mx); s += p[e]; }
    const float is = 1.0f / s;
    for (int e = 0; e < NE; ++e) atomicAdd(&psum[e], p[e] * is);
  }
}

__global__ void moe_scan_aux(const int* __restrict__ counts, int* __restrict__ offs,
                             const float* __restrict__ psum, float* __restrict__ aux_out) {
  if (threadIdx.x == 0) {
    int off = 0; float aux = 0.0f;
    for (int e = 0; e < NE; ++e) {
      offs[e] = off; off += counts[e];
      aux += ((float)counts[e] / (float)TT) * (psum[e] / (float)TT);
    }
    *aux_out = (float)NE * aux;
  }
}

__global__ void moe_scatter(const int* __restrict__ topi, const int* __restrict__ offs,
                            int* __restrict__ cursor, int* __restrict__ stok,
                            int* __restrict__ tslot) {
  int t = blockIdx.x * blockDim.x + threadIdx.x;
  if (t >= TT) return;
  for (int j = 0; j < 2; ++j) {
    const int e = topi[t * 2 + j];
    const int pos = atomicAdd(&cursor[e], 1);
    const int slot = offs[e] + pos;
    stok[slot] = t;
    tslot[t * 2 + j] = slot;
  }
}

// ---------------- fused gate+up+SiLU GEMM (shared or routed) ----------------
// Block: 256 thr / 8 waves; tile M=128, N=64; LDS double-buffered async stages.
// GATHER=false -> dense rows [0,fixed_rows) of xb; GATHER=true -> expert rows.
template <bool GATHER>
__global__ __launch_bounds__(256)
void moe_gateup(const bf16_t* __restrict__ xb, const bf16_t* __restrict__ wg_all,
                const bf16_t* __restrict__ wu_all, const int* __restrict__ stok,
                const int* __restrict__ counts, const int* __restrict__ offs,
                bf16_t* __restrict__ h, int fixed_rows) {
  __shared__ bf16_t sA[2][128 * ROWPAD];   // 128 rows of A
  __shared__ bf16_t sB[2][128 * ROWPAD];   // rows 0-63 gate, 64-127 up
  const int e = blockIdx.x;
  int rows = fixed_rows, base = 0;
  if (GATHER) { rows = counts[e]; base = offs[e]; }
  const int mt0 = blockIdx.y * 128;
  if (mt0 >= rows) return;
  const int n0 = blockIdx.z * 64;
  const bf16_t* wg = wg_all + (size_t)e * DM * DF;
  const bf16_t* wu = wu_all + (size_t)e * DM * DF;
  const int tid = threadIdx.x;
  const unsigned aB0 = lds_off(&sA[0][0]), aB1 = lds_off(&sA[1][0]);
  const unsigned bB0 = lds_off(&sB[0][0]), bB1 = lds_off(&sB[1][0]);
  const int cr = tid >> 2, cc = tid & 3;   // per-thread chunk coords (64 rows x 4 chunks)

  // resolve the two gathered A-row tokens once (fixed per block)
  int tokA0, tokA1;
  if (GATHER) {
    tokA0 = stok[base + ((mt0 + cr      < rows) ? (mt0 + cr)      : (rows - 1))];
    tokA1 = stok[base + ((mt0 + 64 + cr < rows) ? (mt0 + 64 + cr) : (rows - 1))];
  } else {
    tokA0 = mt0 + cr;
    tokA1 = mt0 + 64 + cr;
  }

  auto stage = [&](int buf, int k0) {
    const unsigned ab = buf ? aB1 : aB0;
    const unsigned bb = buf ? bB1 : bB0;
    async_cp16(ab + (unsigned)(cr * ROWPAD + cc * 8) * 2,
               xb + (size_t)tokA0 * DM + k0 + cc * 8);
    async_cp16(ab + (unsigned)((64 + cr) * ROWPAD + cc * 8) * 2,
               xb + (size_t)tokA1 * DM + k0 + cc * 8);
    async_cp16(bb + (unsigned)(cr * ROWPAD + cc * 8) * 2,          // gate rows 0-63
               wg + (size_t)(n0 + cr) * DM + k0 + cc * 8);
    async_cp16(bb + (unsigned)((64 + cr) * ROWPAD + cc * 8) * 2,   // up rows 64-127
               wu + (size_t)(n0 + cr) * DM + k0 + cc * 8);
  };

  const int lane = tid & 31, wave = tid >> 5;
  const int lrow = lane & 15;
  v8f cg[4] = {}, cu[4] = {};
  stage(0, 0);
  const int NIT = DM / KSTEP;                   // 32
  for (int it = 0; it < NIT; ++it) {
    if (it + 1 < NIT) { stage((it + 1) & 1, (it + 1) * KSTEP); WAIT_ASYNC(4); }
    else               WAIT_ASYNC(0);
    __syncthreads();
    const bf16_t* A = &sA[it & 1][0];
    const bf16_t* B = &sB[it & 1][0];
    // issue ALL LDS reads, fence the scheduler, then run the WMMA chain so the
    // waitcnt pass emits graduated s_wait_dscnt instead of full drains
    const v16bf a = lds_a(A, wave * 16 + lrow, lane);
    v16bf bg[4], bu[4];
#pragma unroll
    for (int j = 0; j < 4; ++j) {
      bg[j] = lds_b(B, j * 16 + lrow, lane);
      bu[j] = lds_b(B, 64 + j * 16 + lrow, lane);
    }
    __builtin_amdgcn_sched_barrier(0);
#pragma unroll
    for (int j = 0; j < 4; ++j) {
      cg[j] = wmma_bf16(a, bg[j], cg[j]);
      cu[j] = wmma_bf16(a, bu[j], cu[j]);
    }
    __syncthreads();
  }
  const int half = lane >> 4;
#pragma unroll
  for (int j = 0; j < 4; ++j)
#pragma unroll
    for (int i = 0; i < 8; ++i) {
      const int r = mt0 + wave * 16 + i + half * 8;
      if (r < rows)
        h[(size_t)(base + r) * DF + n0 + j * 16 + lrow] = (bf16_t)(silu(cg[j][i]) * cu[j][i]);
    }
}

// ---------------- down-projection GEMM (shared or routed) ----------------
// out[base+r, DM] = h[base+r, :DF] @ Wd ; ROUTED=false -> dense fixed_rows.
template <bool ROUTED>
__global__ __launch_bounds__(256)
void moe_down(const bf16_t* __restrict__ h, const bf16_t* __restrict__ wd_all,
              const int* __restrict__ counts, const int* __restrict__ offs,
              float* __restrict__ out, int fixed_rows) {
  __shared__ bf16_t sA[2][128 * ROWPAD];
  __shared__ bf16_t sB[2][64 * ROWPAD];
  const int e = blockIdx.x;
  int rows = fixed_rows, base = 0;
  if (ROUTED) { rows = counts[e]; base = offs[e]; }
  const int mt0 = blockIdx.y * 128;
  if (mt0 >= rows) return;
  const int n0 = blockIdx.z * 64;
  const bf16_t* wd = wd_all + (size_t)e * DM * DF;   // [DM rows, DF]
  const int tid = threadIdx.x;
  const unsigned aB0 = lds_off(&sA[0][0]), aB1 = lds_off(&sA[1][0]);
  const unsigned bB0 = lds_off(&sB[0][0]), bB1 = lds_off(&sB[1][0]);
  const int cr = tid >> 2, cc = tid & 3;
  const int ar0 = base + ((mt0 + cr      < rows) ? (mt0 + cr)      : (rows - 1));
  const int ar1 = base + ((mt0 + 64 + cr < rows) ? (mt0 + 64 + cr) : (rows - 1));

  auto stage = [&](int buf, int k0) {
    const unsigned ab = buf ? aB1 : aB0;
    const unsigned bb = buf ? bB1 : bB0;
    async_cp16(ab + (unsigned)(cr * ROWPAD + cc * 8) * 2,
               h + (size_t)ar0 * DF + k0 + cc * 8);
    async_cp16(ab + (unsigned)((64 + cr) * ROWPAD + cc * 8) * 2,
               h + (size_t)ar1 * DF + k0 + cc * 8);
    async_cp16(bb + (unsigned)(cr * ROWPAD + cc * 8) * 2,   // B rows 0-63
               wd + (size_t)(n0 + cr) * DF + k0 + cc * 8);
  };

  const int lane = tid & 31, wave = tid >> 5;
  const int lrow = lane & 15;
  v8f c4[4] = {};
  stage(0, 0);
  const int NIT = DF / KSTEP;                    // 64
  for (int it = 0; it < NIT; ++it) {
    if (it + 1 < NIT) { stage((it + 1) & 1, (it + 1) * KSTEP); WAIT_ASYNC(3); }
    else               WAIT_ASYNC(0);
    __syncthreads();
    const bf16_t* A = &sA[it & 1][0];
    const bf16_t* B = &sB[it & 1][0];
    const v16bf a = lds_a(A, wave * 16 + lrow, lane);
    v16bf bb4[4];
#pragma unroll
    for (int j = 0; j < 4; ++j) bb4[j] = lds_b(B, j * 16 + lrow, lane);
    __builtin_amdgcn_sched_barrier(0);
#pragma unroll
    for (int j = 0; j < 4; ++j) c4[j] = wmma_bf16(a, bb4[j], c4[j]);
    __syncthreads();
  }
  const int half = lane >> 4;
#pragma unroll
  for (int j = 0; j < 4; ++j)
#pragma unroll
    for (int i = 0; i < 8; ++i) {
      const int r = mt0 + wave * 16 + i + half * 8;
      if (r < rows)
        out[(size_t)(base + r) * DM + n0 + j * 16 + lrow] = c4[j][i];
    }
}

// ---------------- final deterministic combine ----------------
__global__ void moe_combine(const float* __restrict__ sout, const float* __restrict__ eout,
                            const float* __restrict__ topw, const int* __restrict__ tslot,
                            float* __restrict__ out) {
  const size_t gid = (size_t)blockIdx.x * blockDim.x + threadIdx.x;
  if (gid >= (size_t)TT * DM) return;
  const int t = (int)(gid / DM);
  const int d = (int)(gid % DM);
  const float w0 = topw[t * 2], w1 = topw[t * 2 + 1];
  const int s0 = tslot[t * 2], s1 = tslot[t * 2 + 1];
  out[gid] = sout[gid] + w0 * eout[(size_t)s0 * DM + d] + w1 * eout[(size_t)s1 * DM + d];
}

// ---------------- host orchestration ----------------
extern "C" void kernel_launch(void* const* d_in, const int* in_sizes, int n_in,
                              void* d_out, int out_size, void* d_ws, size_t ws_size,
                              hipStream_t stream) {
  const float* x   = (const float*)d_in[0];
  const float* swg = (const float*)d_in[1];
  const float* swu = (const float*)d_in[2];
  const float* swd = (const float*)d_in[3];
  const float* ewg = (const float*)d_in[4];
  const float* ewu = (const float*)d_in[5];
  const float* ewd = (const float*)d_in[6];
  const float* rw  = (const float*)d_in[7];
  float* out = (float*)d_out;

  // workspace carve-up (~213 MB total)
  char* ws = (char*)d_ws;
  size_t off = 0;
  auto take = [&](size_t bytes) -> void* {
    void* p = ws + off;
    off = (off + bytes + 255) & ~(size_t)255;
    return p;
  };
  bf16_t* xb    = (bf16_t*)take((size_t)TT * DM * 2);
  bf16_t* swg_t = (bf16_t*)take((size_t)DM * DF * 2);
  bf16_t* swu_t = (bf16_t*)take((size_t)DM * DF * 2);
  bf16_t* swd_t = (bf16_t*)take((size_t)DM * DF * 2);
  bf16_t* ewg_t = (bf16_t*)take((size_t)NE * DM * DF * 2);
  bf16_t* ewu_t = (bf16_t*)take((size_t)NE * DM * DF * 2);
  bf16_t* ewd_t = (bf16_t*)take((size_t)NE * DM * DF * 2);
  bf16_t* hsh   = (bf16_t*)take((size_t)TT * DF * 2);
  float*  sout  = (float*)take((size_t)TT * DM * 4);
  bf16_t* hex   = (bf16_t*)take((size_t)NSLOT * DF * 2);
  float*  eout  = (float*)take((size_t)NSLOT * DM * 4);
  int*    topi  = (int*)take((size_t)TT * 2 * 4);
  float*  topw  = (float*)take((size_t)TT * 2 * 4);
  int*    tslot = (int*)take((size_t)TT * 2 * 4);
  int*    stok  = (int*)take((size_t)NSLOT * 4);
  int*    ctrl  = (int*)take(256);
  int*   counts = ctrl;                 // [8]
  int*   cursor = ctrl + 8;             // [8]
  int*   offs   = ctrl + 16;            // [8]
  float* psum   = (float*)(ctrl + 24);  // [8]

  hipMemsetAsync(ctrl, 0, 256, stream);

  // 1) fp32 -> bf16 activation copy
  moe_cvt_x<<<(TT * DM + 255) / 256, 256, 0, stream>>>(x, xb, TT * DM);

  // 2) weight transpose + convert (dst is [N,K] bf16)
  moe_transpose<<<dim3(DM / 32, DF / 32, 1),  256, 0, stream>>>(swg, swg_t, DM, DF);
  moe_transpose<<<dim3(DM / 32, DF / 32, 1),  256, 0, stream>>>(swu, swu_t, DM, DF);
  moe_transpose<<<dim3(DF / 32, DM / 32, 1),  256, 0, stream>>>(swd, swd_t, DF, DM);
  moe_transpose<<<dim3(DM / 32, DF / 32, NE), 256, 0, stream>>>(ewg, ewg_t, DM, DF);
  moe_transpose<<<dim3(DM / 32, DF / 32, NE), 256, 0, stream>>>(ewu, ewu_t, DM, DF);
  moe_transpose<<<dim3(DF / 32, DM / 32, NE), 256, 0, stream>>>(ewd, ewd_t, DF, DM);

  // 3) routing + aux loss + dispatch
  moe_router<<<TT / 8, 256, 0, stream>>>(x, rw, topi, topw, counts, psum);
  moe_scan_aux<<<1, 32, 0, stream>>>(counts, offs, psum, out + (size_t)TT * DM);
  moe_scatter<<<TT / 256, 256, 0, stream>>>(topi, offs, cursor, stok, tslot);

  // 4) shared expert (dense path)
  moe_gateup<false><<<dim3(1, TT / 128, DF / 64), 256, 0, stream>>>(
      xb, swg_t, swu_t, nullptr, nullptr, nullptr, hsh, TT);
  moe_down<false><<<dim3(1, TT / 128, DM / 64), 256, 0, stream>>>(
      hsh, swd_t, nullptr, nullptr, sout, TT);

  // 5) routed experts (gathered rows; worst case 64 m-tiles per expert)
  moe_gateup<true><<<dim3(NE, NSLOT / 128, DF / 64), 256, 0, stream>>>(
      xb, ewg_t, ewu_t, stok, counts, offs, hex, 0);
  moe_down<true><<<dim3(NE, NSLOT / 128, DM / 64), 256, 0, stream>>>(
      hex, ewd_t, counts, offs, eout, 0);

  // 6) deterministic combine into d_out
  moe_combine<<<(TT * DM + 255) / 256, 256, 0, stream>>>(sout, eout, topw, tslot, out);
}
